// PA_STGAT_4896262717767
// MI455X (gfx1250) — matrix operand order repaired and checked by reading, not compiled, not run on hardware
//
#include <hip/hip_runtime.h>
#include <hip/hip_bf16.h>

#define N_NODES 10000
#define T_STEPS 12
#define F_IN    2
#define EMB_D   64
#define HID_D   64
#define HEADS   4
#define E_EDGES 320000
#define E_TOT   (E_EDGES + N_NODES)
#define NPAD    10016            // N + 16, a multiple of 16
#define NEG_SLOPE 0.2f

typedef __attribute__((ext_vector_type(16))) _Float16 v16h;
typedef __attribute__((ext_vector_type(8)))  _Float16 v8h;
typedef __attribute__((ext_vector_type(8)))  float    v8f;

// ---------- generic fills / converts ----------
__global__ void stgat_fill_u32(unsigned* __restrict p, unsigned v, int n) {
    int i = blockIdx.x * blockDim.x + threadIdx.x;
    if (i < n) p[i] = v;
}
__global__ void stgat_cvt_f16(_Float16* __restrict d, const float* __restrict s, int n) {
    int i = blockIdx.x * blockDim.x + threadIdx.x;
    if (i < n) d[i] = (_Float16)s[i];
}
// W_gat [64][256] (K x Nc) -> BT [256][64]
__global__ void stgat_wgat_t(_Float16* __restrict d, const float* __restrict s) {
    int i = blockIdx.x * blockDim.x + threadIdx.x;
    if (i < 256 * 64) {
        int n = i >> 6, k = i & 63;
        d[i] = (_Float16)s[k * 256 + n];
    }
}

__device__ __forceinline__ void edge_sd(const int* __restrict ei, int e, int& s, int& d) {
    if (e < E_EDGES) { s = ei[e]; d = ei[E_EDGES + e]; }
    else             { s = d = e - E_EDGES; }           // appended self loops
}

// ---------- CSR build (edge_index is fixed per launch) ----------
__global__ void stgat_deg(const int* __restrict ei, int* __restrict deg) {
    int e = blockIdx.x * blockDim.x + threadIdx.x;
    if (e >= E_TOT) return;
    int s, d; edge_sd(ei, e, s, d);
    atomicAdd(&deg[d], 1);
}
// single-block exclusive scan over deg[N_NODES] -> rowptr[N_NODES+1], cursor copy
__global__ __launch_bounds__(256) void stgat_scan(const int* __restrict deg,
                                                  int* __restrict rowptr,
                                                  int* __restrict cursor) {
    __shared__ int ssum[256];
    const int tid = threadIdx.x;
    const int CH = (N_NODES + 255) / 256;   // 40
    const int base = tid * CH;
    int s = 0;
    for (int j = 0; j < CH; ++j) { int idx = base + j; if (idx < N_NODES) s += deg[idx]; }
    ssum[tid] = s;
    __syncthreads();
    for (int off = 1; off < 256; off <<= 1) {       // inclusive Hillis-Steele
        int v = ssum[tid];
        int add = (tid >= off) ? ssum[tid - off] : 0;
        __syncthreads();
        ssum[tid] = v + add;
        __syncthreads();
    }
    int run = (tid == 0) ? 0 : ssum[tid - 1];       // exclusive chunk prefix
    for (int j = 0; j < CH; ++j) {
        int idx = base + j;
        if (idx < N_NODES) {
            rowptr[idx] = run;
            cursor[idx] = run;
            run += deg[idx];
        }
    }
    if (tid == 255) rowptr[N_NODES] = run;          // == E_TOT
}
__global__ void stgat_scatter(const int* __restrict ei, int* __restrict cursor,
                              int* __restrict csr_src) {
    int e = blockIdx.x * blockDim.x + threadIdx.x;
    if (e >= E_TOT) return;
    int s, d; edge_sd(ei, e, s, d);
    int pos = atomicAdd(&cursor[d], 1);
    csr_src[pos] = s;
}

// ---------- embedding: emb = relu(x_t @ W_emb + b_emb), f16 out, pad rows = 0 ----------
__global__ void stgat_embed(const float* __restrict x, const float* __restrict Wemb,
                            const float* __restrict bemb, _Float16* __restrict emb, int t) {
    int i = blockIdx.x * blockDim.x + threadIdx.x;
    if (i >= NPAD * EMB_D) return;
    int n = i >> 6, j = i & 63;
    _Float16 v = (_Float16)0.f;
    if (n < N_NODES) {
        float x0 = x[n * T_STEPS * F_IN + t * F_IN + 0];
        float x1 = x[n * T_STEPS * F_IN + t * F_IN + 1];
        float e = x0 * Wemb[j] + x1 * Wemb[EMB_D + j] + bemb[j];
        v = (_Float16)(e > 0.f ? e : 0.f);
    }
    emb[i] = v;
}

// ---------- WMMA GEMM: C[M x Nc] = A[M x K] * BT[Nc x K]^T, f16 in, f32 out ----------
// one wave per block; wave computes a 16x64 strip (A fragment reused for 4 WMMAs)
// grid.x = M/16, grid.y = Nc/64
template <int K, int LDA, int LDBT, int LDC>
__global__ __launch_bounds__(32) void stgat_gemm(const _Float16* __restrict A,
                                                 const _Float16* __restrict BT,
                                                 float* __restrict C) {
    const int lane = threadIdx.x & 31;
    const int m0   = blockIdx.x * 16;
    const int n0   = blockIdx.y * 64;
    const int mrow = m0 + (lane & 15);
    const int kbA  = (lane >> 4) * 16;   // A: lane half -> K 0-15 / 16-31
    const int kbB  = (lane >> 4) * 16;   // B: lane half -> K 0-15 / 16-31

    const _Float16* __restrict Arow = A  + (size_t)mrow * LDA;
    const _Float16* __restrict B0   = BT + (size_t)(n0 + (lane & 15)) * LDBT;

    v8f acc0 = {}, acc1 = {}, acc2 = {}, acc3 = {};
#pragma unroll
    for (int k0 = 0; k0 < K; k0 += 32) {
        v8h alo = *(const v8h*)(Arow + k0 + kbA);
        v8h ahi = *(const v8h*)(Arow + k0 + kbA + 8);
        v16h a = __builtin_shufflevector(alo, ahi, 0, 1, 2, 3, 4, 5, 6, 7,
                                         8, 9, 10, 11, 12, 13, 14, 15);
        v16h b0 = *(const v16h*)(B0 + (size_t) 0 * LDBT + k0 + kbB);
        v16h b1 = *(const v16h*)(B0 + (size_t)16 * LDBT + k0 + kbB);
        v16h b2 = *(const v16h*)(B0 + (size_t)32 * LDBT + k0 + kbB);
        v16h b3 = *(const v16h*)(B0 + (size_t)48 * LDBT + k0 + kbB);
        acc0 = __builtin_amdgcn_wmma_f32_16x16x32_f16(false, a, false, b0, (short)0, acc0, false, false);
        acc1 = __builtin_amdgcn_wmma_f32_16x16x32_f16(false, a, false, b1, (short)0, acc1, false, false);
        acc2 = __builtin_amdgcn_wmma_f32_16x16x32_f16(false, a, false, b2, (short)0, acc2, false, false);
        acc3 = __builtin_amdgcn_wmma_f32_16x16x32_f16(false, a, false, b3, (short)0, acc3, false, false);
    }
    const int rbase = m0 + (lane >> 4) * 8;
    const int c0 = n0 + (lane & 15);
#pragma unroll
    for (int e = 0; e < 8; ++e) {
        float* crow = C + (size_t)(rbase + e) * LDC + c0;
        crow[0]  = acc0[e];
        crow[16] = acc1[e];
        crow[32] = acc2[e];
        crow[48] = acc3[e];
    }
}

// ---------- attention per-node dots: a_s[n][h], a_d[n][h] ----------
__global__ void stgat_attdot(const float* __restrict hfeat, const float* __restrict att_src,
                             const float* __restrict att_dst, float* __restrict a_s,
                             float* __restrict a_d) {
    int i = blockIdx.x * blockDim.x + threadIdx.x;
    if (i >= N_NODES * HEADS) return;
    int n = i >> 2, h = i & 3;
    const float* hp = hfeat + (size_t)n * 256 + h * 64;
    const float* as = att_src + h * 64;
    const float* ad = att_dst + h * 64;
    float sa = 0.f, sd = 0.f;
#pragma unroll 8
    for (int d = 0; d < 64; ++d) { float v = hp[d]; sa += v * as[d]; sd += v * ad[d]; }
    a_s[i] = sa; a_d[i] = sd;
}

// ---------- segment softmax over in-edges, no atomics (one thread per (n,h)) ----------
// alpha stored at CSR position: ebuf[p*4 + h]
__global__ void stgat_softmax(const int* __restrict rowptr, const int* __restrict csr_src,
                              const float* __restrict a_s, const float* __restrict a_d,
                              float* __restrict ebuf) {
    int i = blockIdx.x * blockDim.x + threadIdx.x;
    if (i >= N_NODES * HEADS) return;
    int n = i >> 2, h = i & 3;
    int p0 = rowptr[n], p1 = rowptr[n + 1];
    float ad = a_d[n * 4 + h];
    float m = -INFINITY;                               // p1 > p0 (self loop) -> finite
    for (int p = p0; p < p1; ++p) {
        float a = a_s[csr_src[p] * 4 + h] + ad;
        a = (a > 0.f) ? a : NEG_SLOPE * a;
        ebuf[p * 4 + h] = a;
        m = fmaxf(m, a);
    }
    float den = 0.f;
    for (int p = p0; p < p1; ++p) {
        float ex = __expf(ebuf[p * 4 + h] - m);
        ebuf[p * 4 + h] = ex;
        den += ex;
    }
    float inv = 1.f / (den + 1e-16f);
    for (int p = p0; p < p1; ++p) ebuf[p * 4 + h] *= inv;
}

// ---------- aggregation, no atomics: one wave per (n,h), lane owns 2 channels ----------
// writes gat16 = (sum alpha*hfeat[src]) + b_gat directly in f16
__global__ __launch_bounds__(256) void stgat_aggregate(const int* __restrict rowptr,
                                                       const int* __restrict csr_src,
                                                       const float* __restrict ebuf,
                                                       const float* __restrict hfeat,
                                                       const float* __restrict bgat,
                                                       _Float16* __restrict gat16) {
    int wave = (blockIdx.x * blockDim.x + threadIdx.x) >> 5;
    int lane = threadIdx.x & 31;
    if (wave >= N_NODES * HEADS) return;
    int n = wave >> 2, h = wave & 3;
    int p0 = rowptr[n], p1 = rowptr[n + 1];
    float acc0 = 0.f, acc1 = 0.f;
    for (int p = p0; p < p1; ++p) {
        float alpha = ebuf[p * 4 + h];
        const float* hp = hfeat + (size_t)csr_src[p] * 256 + h * 64;
        acc0 += alpha * hp[lane];
        acc1 += alpha * hp[lane + 32];
    }
    size_t o = (size_t)n * 256 + h * 64;
    gat16[o + lane]      = (_Float16)(acc0 + bgat[h * 64 + lane]);
    gat16[o + lane + 32] = (_Float16)(acc1 + bgat[h * 64 + lane + 32]);
}

// ---------- GRU gates (torch order r,z,n) ----------
__global__ void stgat_gru(const float* __restrict gi, const float* __restrict gh,
                          const float* __restrict bih, const float* __restrict bhh,
                          float* __restrict h, _Float16* __restrict h16) {
    int i = blockIdx.x * blockDim.x + threadIdx.x;
    if (i >= NPAD * HID_D) return;
    int n = i >> 6, d = i & 63;
    float hv = 0.f;
    if (n < N_NODES) {
        const float* gip = gi + (size_t)n * 192;
        const float* ghp = gh + (size_t)n * 192;
        float ir  = gip[d]        + bih[d];
        float iz  = gip[64 + d]   + bih[64 + d];
        float in_ = gip[128 + d]  + bih[128 + d];
        float hr  = ghp[d]        + bhh[d];
        float hz  = ghp[64 + d]   + bhh[64 + d];
        float hn  = ghp[128 + d]  + bhh[128 + d];
        float r  = 1.f / (1.f + __expf(-(ir + hr)));
        float z  = 1.f / (1.f + __expf(-(iz + hz)));
        float nn = tanhf(in_ + r * hn);
        hv = (1.f - z) * nn + z * h[i];
    }
    h[i] = hv;
    h16[i] = (_Float16)hv;
}

// ---------- output head ----------
__global__ void stgat_out(const float* __restrict h, const float* __restrict Wout,
                          const float* __restrict bout, float* __restrict out) {
    int n = blockIdx.x * blockDim.x + threadIdx.x;
    if (n >= N_NODES) return;
    const float* hp = h + (size_t)n * HID_D;
    float s = bout[0];
#pragma unroll 8
    for (int d = 0; d < HID_D; ++d) s += hp[d] * Wout[d];
    out[n] = s;
}

extern "C" void kernel_launch(void* const* d_in, const int* in_sizes, int n_in,
                              void* d_out, int out_size, void* d_ws, size_t ws_size,
                              hipStream_t stream) {
    const float* x       = (const float*)d_in[0];
    const int*   ei      = (const int*)  d_in[1];
    const float* W_emb   = (const float*)d_in[2];
    const float* b_emb   = (const float*)d_in[3];
    const float* W_gat   = (const float*)d_in[4];
    const float* att_src = (const float*)d_in[5];
    const float* att_dst = (const float*)d_in[6];
    const float* b_gat   = (const float*)d_in[7];
    const float* W_ih    = (const float*)d_in[8];
    const float* W_hh    = (const float*)d_in[9];
    const float* b_ih    = (const float*)d_in[10];
    const float* b_hh    = (const float*)d_in[11];
    const float* W_out   = (const float*)d_in[12];
    const float* b_out   = (const float*)d_in[13];
    float* out = (float*)d_out;

    char* p = (char*)d_ws;
    auto carve = [&](size_t bytes) -> void* {
        void* r = (void*)p;
        p += (bytes + 255) & ~(size_t)255;
        return r;
    };
    _Float16* wgatT  = (_Float16*)carve(256 * 64 * 2);               // [256][64]  = W_gat^T
    _Float16* wih16  = (_Float16*)carve(192 * 256 * 2);              // [192][256] = W_ih (already BT)
    _Float16* whh16  = (_Float16*)carve(192 * 64 * 2);               // [192][64]  = W_hh (already BT)
    _Float16* emb16  = (_Float16*)carve((size_t)NPAD * 64 * 2);
    float*    hfeat  = (float*)   carve((size_t)NPAD * 256 * 4);
    float*    a_s    = (float*)   carve((size_t)N_NODES * 4 * 4);
    float*    a_d    = (float*)   carve((size_t)N_NODES * 4 * 4);
    float*    ebuf   = (float*)   carve((size_t)E_TOT * 4 * 4);
    _Float16* gat16  = (_Float16*)carve((size_t)NPAD * 256 * 2);
    float*    gi     = (float*)   carve((size_t)NPAD * 192 * 4);
    float*    gh     = (float*)   carve((size_t)NPAD * 192 * 4);
    float*    hbuf   = (float*)   carve((size_t)NPAD * 64 * 4);
    _Float16* h16    = (_Float16*)carve((size_t)NPAD * 64 * 2);
    int*      deg    = (int*)     carve((size_t)N_NODES * 4);
    int*      rowptr = (int*)     carve((size_t)(N_NODES + 1) * 4);
    int*      cursor = (int*)     carve((size_t)N_NODES * 4);
    int*      csrsrc = (int*)     carve((size_t)E_TOT * 4);

    auto nb = [](int n) { return dim3((unsigned)((n + 255) / 256)); };

    // ---- per-launch prep: f16 weights, CSR by dst, h0 = 0, gat16 pad rows = 0 ----
    stgat_wgat_t <<<nb(256 * 64), 256, 0, stream>>>(wgatT, W_gat);
    stgat_cvt_f16<<<nb(192 * 256), 256, 0, stream>>>(wih16, W_ih, 192 * 256);
    stgat_cvt_f16<<<nb(192 * 64), 256, 0, stream>>>(whh16, W_hh, 192 * 64);
    stgat_fill_u32<<<nb(NPAD * 64), 256, 0, stream>>>((unsigned*)hbuf, 0u, NPAD * 64);
    stgat_fill_u32<<<nb(NPAD * 32), 256, 0, stream>>>((unsigned*)h16, 0u, NPAD * 32);
    stgat_fill_u32<<<nb((NPAD - N_NODES) * 128), 256, 0, stream>>>(
        (unsigned*)(gat16 + (size_t)N_NODES * 256), 0u, (NPAD - N_NODES) * 128);
    stgat_fill_u32<<<nb(N_NODES), 256, 0, stream>>>((unsigned*)deg, 0u, N_NODES);
    stgat_deg    <<<nb(E_TOT), 256, 0, stream>>>(ei, deg);
    stgat_scan   <<<1, 256, 0, stream>>>(deg, rowptr, cursor);
    stgat_scatter<<<nb(E_TOT), 256, 0, stream>>>(ei, cursor, csrsrc);

    for (int t = 0; t < T_STEPS; ++t) {
        stgat_embed<<<nb(NPAD * 64), 256, 0, stream>>>(x, W_emb, b_emb, emb16, t);

        // hfeat = emb @ W_gat   [NPAD x 64] * [64 x 256]
        stgat_gemm<64, 64, 64, 256><<<dim3(NPAD / 16, 4), 32, 0, stream>>>(emb16, wgatT, hfeat);

        stgat_attdot<<<nb(N_NODES * 4), 256, 0, stream>>>(hfeat, att_src, att_dst, a_s, a_d);
        stgat_softmax<<<nb(N_NODES * 4), 256, 0, stream>>>(rowptr, csrsrc, a_s, a_d, ebuf);
        stgat_aggregate<<<nb(N_NODES * 4 * 32), 256, 0, stream>>>(rowptr, csrsrc, ebuf,
                                                                  hfeat, b_gat, gat16);

        // gi = gat @ W_ih^T   [NPAD x 256] * [256 x 192]
        stgat_gemm<256, 256, 256, 192><<<dim3(NPAD / 16, 3), 32, 0, stream>>>(gat16, wih16, gi);
        // gh = h @ W_hh^T     [NPAD x 64] * [64 x 192]
        stgat_gemm<64, 64, 64, 192><<<dim3(NPAD / 16, 3), 32, 0, stream>>>(h16, whh16, gh);

        stgat_gru<<<nb(NPAD * 64), 256, 0, stream>>>(gi, gh, b_ih, b_hh, hbuf, h16);
    }

    stgat_out<<<nb(N_NODES), 256, 0, stream>>>(hbuf, W_out, b_out, out);
}
